// ComplexEncoder_27977416966201
// MI455X (gfx1250) — matrix operand order repaired and checked by reading, not compiled, not run on hardware
//
#include <hip/hip_runtime.h>
#include <hip/hip_bf16.h>

typedef unsigned short u16;
typedef __attribute__((ext_vector_type(16))) __bf16 v16bf;
typedef __attribute__((ext_vector_type(8)))  float  v8f;
typedef __attribute__((ext_vector_type(4)))  unsigned int u32x4;

// ---------------- constants ----------------
#define BB   4
#define SS   1024
#define HH   1024
#define NH   16
#define DH   64
#define FFD  4096
#define BS   (BB*SS)          // 4096 token rows

// ---------------- helpers ----------------
__device__ __forceinline__ u16 f2bf(float x) {
    unsigned int u = __float_as_uint(x);
    unsigned int r = u + 0x7FFFu + ((u >> 16) & 1u);   // round-to-nearest-even
    return (u16)(r >> 16);
}

union FragU { v16bf v; u32x4 q[2]; };

// A-operand fragment (16x32 bf16, M x K):
// lane m (0-15): elems 0-7 = K0-7, elems 8-15 = K16-23
// lane m+16    : elems 0-7 = K8-15, elems 8-15 = K24-31
__device__ __forceinline__ v16bf load_a_frag(const u16* base, size_t row, int ldk, int k0, int lane) {
    int klo = (lane & 16) ? 8 : 0;
    const u16* p = base + row * (size_t)ldk + k0 + klo;
    FragU f;
    f.q[0] = *(const u32x4*)(p);
    f.q[1] = *(const u32x4*)(p + 16);
    return f.v;
}

// B-operand fragment (32x16 bf16, K x N), sourced from B^T stored [N][K]:
// lane n (0-15): elems 0-7 = K0-7, elems 8-15 = K16-23 of column n; lane n+16: K8-15, K24-31
__device__ __forceinline__ v16bf load_b_frag(const u16* base, size_t row, int ldk, int k0, int lane) {
    int koff = (lane & 16) ? 8 : 0;
    const u16* p = base + row * (size_t)ldk + k0 + koff;
    FragU f;
    f.q[0] = *(const u32x4*)(p);
    f.q[1] = *(const u32x4*)(p + 16);
    return f.v;
}

__device__ __forceinline__ v8f wmma_bf16(v16bf a, v16bf b, v8f c) {
    return __builtin_amdgcn_wmma_f32_16x16x32_bf16(
        /*neg_a=*/false, a, /*neg_b=*/false, b,
        /*c_mod=*/(short)0, c, /*reuse_a=*/false, /*reuse_b=*/false);
}

__device__ __forceinline__ v8f vzero() {
    v8f z;
#pragma unroll
    for (int i = 0; i < 8; ++i) z[i] = 0.0f;
    return z;
}

// ---------------- elementwise / layout kernels ----------------
__global__ void k_f2bf(const float* __restrict__ in, u16* __restrict__ out, int n) {
    int i = blockIdx.x * 256 + threadIdx.x;
    if (i < n) out[i] = f2bf(in[i]);
}

// in [K][C] f32  ->  out [C][K] bf16 (transposed weight for B-operand)
__global__ void k_transpose_bf16(const float* __restrict__ in, u16* __restrict__ out, int K, int C) {
    int i = blockIdx.x * 256 + threadIdx.x;
    if (i >= K * C) return;
    int c = i / K, k = i - c * K;
    out[(size_t)c * K + k] = f2bf(in[(size_t)k * C + c]);
}

// per-head weight [N][H][D] f32 -> [C=N*D][H] bf16
__global__ void k_headw_t(const float* __restrict__ in, u16* __restrict__ out) {
    int i = blockIdx.x * 256 + threadIdx.x;
    if (i >= 1024 * HH) return;
    int c = i / HH, h = i - c * HH;
    int n = c >> 6, d = c & 63;
    out[(size_t)c * HH + h] = f2bf(in[((size_t)n * HH + h) * DH + d]);
}

// q epilogue: out = bf16( sign * (tmp*scale + pos[n,s,d]) )   (bias already in tmp)
__global__ void k_epi_q(const float* __restrict__ tmp, const float* __restrict__ pos,
                        u16* __restrict__ out, float scale, int negate) {
    int i = blockIdx.x * 256 + threadIdx.x;
    if (i >= BS * 1024) return;
    int bs = i >> 10, c = i & 1023;
    int s = bs & (SS - 1);
    int n = c >> 6, d = c & 63;
    float v = tmp[i] * scale + pos[((size_t)n * SS + s) * DH + d];
    if (negate) v = -v;
    out[i] = f2bf(v);
}

__global__ void k_epi_k(const float* __restrict__ tmp, const float* __restrict__ pos,
                        u16* __restrict__ out) {
    int i = blockIdx.x * 256 + threadIdx.x;
    if (i >= BS * 1024) return;
    int bs = i >> 10, c = i & 1023;
    int s = bs & (SS - 1);
    int n = c >> 6, d = c & 63;
    out[i] = f2bf(tmp[i] + pos[((size_t)n * SS + s) * DH + d]);
}

// v epilogue: write V transposed  vT[b][n][d][s] bf16
__global__ void k_epi_v(const float* __restrict__ tmp, u16* __restrict__ vT) {
    int i = blockIdx.x * 256 + threadIdx.x;
    if (i >= BS * 1024) return;
    int bs = i >> 10, c = i & 1023;
    int b = bs >> 10, s = bs & (SS - 1);
    int n = c >> 6, d = c & 63;
    vT[((size_t)((b * NH + n) * DH + d)) * SS + s] = f2bf(tmp[i]);
}

// ---------------- generic WMMA GEMM ----------------
// C[M,N] = A[M,K](bf16,row-major) * B[K,N] given as Bt[N,K](bf16) + bias[N]
// mode 0: f32 out;  mode 1: bf16(relu(out));  mode 2: bf16(out)
__global__ __launch_bounds__(256) void k_gemm(const u16* __restrict__ A, const u16* __restrict__ Bt,
                                              const float* __restrict__ bias, void* __restrict__ out,
                                              int M, int N, int K, int mode) {
    int lane = threadIdx.x;
    int wave = threadIdx.y;
    int m0 = (blockIdx.y * 8 + wave) * 32;
    int n0 = blockIdx.x * 64;
    if (m0 >= M) return;
    int lr = lane & 15;
    int hi = (lane >> 4) & 1;

    v8f acc[2][4];
#pragma unroll
    for (int i = 0; i < 2; ++i)
#pragma unroll
        for (int j = 0; j < 4; ++j) acc[i][j] = vzero();

    for (int k0 = 0; k0 < K; k0 += 32) {
        if (k0 + 32 < K) {
            __builtin_prefetch((const void*)(A  + (size_t)(m0 + lr) * K + k0 + 32), 0, 1);
            __builtin_prefetch((const void*)(Bt + (size_t)(n0 + lr) * K + k0 + 32), 0, 1);
        }
        v16bf a0 = load_a_frag(A, (size_t)(m0 + lr),      K, k0, lane);
        v16bf a1 = load_a_frag(A, (size_t)(m0 + 16 + lr), K, k0, lane);
        v16bf b0 = load_b_frag(Bt, (size_t)(n0 + 0  + lr), K, k0, lane);
        v16bf b1 = load_b_frag(Bt, (size_t)(n0 + 16 + lr), K, k0, lane);
        v16bf b2 = load_b_frag(Bt, (size_t)(n0 + 32 + lr), K, k0, lane);
        v16bf b3 = load_b_frag(Bt, (size_t)(n0 + 48 + lr), K, k0, lane);
        acc[0][0] = wmma_bf16(a0, b0, acc[0][0]);
        acc[0][1] = wmma_bf16(a0, b1, acc[0][1]);
        acc[0][2] = wmma_bf16(a0, b2, acc[0][2]);
        acc[0][3] = wmma_bf16(a0, b3, acc[0][3]);
        acc[1][0] = wmma_bf16(a1, b0, acc[1][0]);
        acc[1][1] = wmma_bf16(a1, b1, acc[1][1]);
        acc[1][2] = wmma_bf16(a1, b2, acc[1][2]);
        acc[1][3] = wmma_bf16(a1, b3, acc[1][3]);
    }

#pragma unroll
    for (int i = 0; i < 2; ++i)
#pragma unroll
        for (int j = 0; j < 4; ++j) {
            int cn = n0 + j * 16 + lr;
            float bi = bias ? bias[cn] : 0.0f;
#pragma unroll
            for (int r = 0; r < 8; ++r) {
                int rm = m0 + i * 16 + r + hi * 8;
                float v = acc[i][j][r] + bi;
                if (mode == 1) {
                    v = fmaxf(v, 0.0f);
                    ((u16*)out)[(size_t)rm * N + cn] = f2bf(v);
                } else if (mode == 2) {
                    ((u16*)out)[(size_t)rm * N + cn] = f2bf(v);
                } else {
                    ((float*)out)[(size_t)rm * N + cn] = v;
                }
            }
        }
}

// ---------------- fused flash attention ----------------
// scores = (qr*kr + (-qi)*ki) * scale, online softmax, O += P*V, out -> heads bf16 [BS][N*D]
__global__ __launch_bounds__(256) void k_flash(const u16* __restrict__ qrb, const u16* __restrict__ qnb,
                                               const u16* __restrict__ krb, const u16* __restrict__ kib,
                                               const u16* __restrict__ vT, u16* __restrict__ headsb,
                                               float scale) {
    __shared__ u16 Pld[8][16][32];     // per-wave P staging (bf16)
    int lane = threadIdx.x, wave = threadIdx.y;
    int wg = blockIdx.x * 8 + wave;    // 4096 waves total
    int mtile = wg & 63;               // S/16
    int head  = (wg >> 6) & 15;
    int b     = wg >> 10;
    int lr = lane & 15;
    int hi = (lane >> 4) & 1;
    int m0 = mtile * 16;
    int cbase = head * DH;

    size_t qrow = (size_t)(b * SS + m0 + lr);
    v16bf qr0 = load_a_frag(qrb, qrow, 1024, cbase + 0,  lane);
    v16bf qr1 = load_a_frag(qrb, qrow, 1024, cbase + 32, lane);
    v16bf qn0 = load_a_frag(qnb, qrow, 1024, cbase + 0,  lane);
    v16bf qn1 = load_a_frag(qnb, qrow, 1024, cbase + 32, lane);

    float m8[8], l8[8];
    v8f o[4];
#pragma unroll
    for (int r = 0; r < 8; ++r) { m8[r] = -3.0e38f; l8[r] = 0.0f; }
#pragma unroll
    for (int j = 0; j < 4; ++j) o[j] = vzero();

    for (int t0 = 0; t0 < SS; t0 += 32) {
        v8f s[2];
#pragma unroll
        for (int sub = 0; sub < 2; ++sub) {
            size_t trow = (size_t)(b * SS + t0 + sub * 16 + lr);
            v16bf kr0 = load_b_frag(krb, trow, 1024, cbase + 0,  lane);
            v16bf kr1 = load_b_frag(krb, trow, 1024, cbase + 32, lane);
            v16bf ki0 = load_b_frag(kib, trow, 1024, cbase + 0,  lane);
            v16bf ki1 = load_b_frag(kib, trow, 1024, cbase + 32, lane);
            v8f c = vzero();
            c = wmma_bf16(qr0, kr0, c);
            c = wmma_bf16(qr1, kr1, c);
            c = wmma_bf16(qn0, ki0, c);   // qn = -qi, so this subtracts qi*ki
            c = wmma_bf16(qn1, ki1, c);
            s[sub] = c;
        }
        // online softmax: rows live per 16-lane half (rows r / r+8)
#pragma unroll
        for (int r = 0; r < 8; ++r) {
            float s0 = s[0][r] * scale, s1 = s[1][r] * scale;
            float mx = fmaxf(s0, s1);
            mx = fmaxf(mx, __shfl_xor(mx, 1, 32));
            mx = fmaxf(mx, __shfl_xor(mx, 2, 32));
            mx = fmaxf(mx, __shfl_xor(mx, 4, 32));
            mx = fmaxf(mx, __shfl_xor(mx, 8, 32));
            float mn = fmaxf(m8[r], mx);
            float al = __expf(m8[r] - mn);
            m8[r] = mn;
            float p0 = __expf(s0 - mn);
            float p1 = __expf(s1 - mn);
            float rs = p0 + p1;
            rs += __shfl_xor(rs, 1, 32);
            rs += __shfl_xor(rs, 2, 32);
            rs += __shfl_xor(rs, 4, 32);
            rs += __shfl_xor(rs, 8, 32);
            l8[r] = l8[r] * al + rs;
            Pld[wave][r + hi * 8][lr]      = f2bf(p0);
            Pld[wave][r + hi * 8][16 + lr] = f2bf(p1);
#pragma unroll
            for (int j = 0; j < 4; ++j) o[j][r] *= al;
        }
        // reload P in A-fragment layout from LDS (same-wave LDS ops are in order)
        v16bf pf = load_a_frag(&Pld[wave][0][0], (size_t)lr, 32, 0, lane);
#pragma unroll
        for (int j = 0; j < 4; ++j) {
            size_t vrow = (size_t)((b * NH + head) * DH + j * 16 + lr);
            v16bf vf = load_b_frag(vT, vrow, SS, t0, lane);
            o[j] = wmma_bf16(pf, vf, o[j]);
        }
    }
#pragma unroll
    for (int j = 0; j < 4; ++j)
#pragma unroll
        for (int r = 0; r < 8; ++r) {
            int srow = m0 + r + hi * 8;
            headsb[((size_t)(b * SS + srow)) * 1024 + cbase + j * 16 + lr] = f2bf(o[j][r] / l8[r]);
        }
}

// ---------------- residual + layernorm ----------------
__global__ __launch_bounds__(256) void k_ln(const float* __restrict__ x, const float* __restrict__ y,
                                            const float* __restrict__ g, const float* __restrict__ be,
                                            float* __restrict__ out, u16* __restrict__ outb, int H) {
    int row = blockIdx.x, tid = threadIdx.x;
    const float* xr = x + (size_t)row * H;
    const float* yr = y + (size_t)row * H;
    float s = 0.0f, ss = 0.0f;
    for (int i = tid; i < H; i += 256) {
        float v = xr[i] + yr[i];
        s += v; ss += v * v;
    }
    __shared__ float rs[256], rq[256];
    rs[tid] = s; rq[tid] = ss;
    __syncthreads();
    for (int o2 = 128; o2 > 0; o2 >>= 1) {
        if (tid < o2) { rs[tid] += rs[tid + o2]; rq[tid] += rq[tid + o2]; }
        __syncthreads();
    }
    float mu = rs[0] / H;
    float var = rq[0] / H - mu * mu;
    float rstd = rsqrtf(var + 1e-5f);
    for (int i = tid; i < H; i += 256) {
        float v = (xr[i] + yr[i] - mu) * rstd * g[i] + be[i];
        out[(size_t)row * H + i] = v;
        if (outb) outb[(size_t)row * H + i] = f2bf(v);
    }
}

// ---------------- launcher ----------------
extern "C" void kernel_launch(void* const* d_in, const int* in_sizes, int n_in,
                              void* d_out, int out_size, void* d_ws, size_t ws_size,
                              hipStream_t stream) {
    (void)in_sizes; (void)n_in; (void)out_size; (void)ws_size;
    const float* x    = (const float*)d_in[0];
    const float* Wq_r = (const float*)d_in[1];
    const float* Wq_i = (const float*)d_in[2];
    const float* bq_r = (const float*)d_in[3];
    const float* bq_i = (const float*)d_in[4];
    const float* Wk_r = (const float*)d_in[5];
    const float* Wk_i = (const float*)d_in[6];
    const float* bk_r = (const float*)d_in[7];
    const float* bk_i = (const float*)d_in[8];
    const float* Wv   = (const float*)d_in[9];
    const float* bv   = (const float*)d_in[10];
    const float* pqr  = (const float*)d_in[11];
    const float* pqi  = (const float*)d_in[12];
    const float* pkr  = (const float*)d_in[13];
    const float* pki  = (const float*)d_in[14];
    const float* Wo   = (const float*)d_in[15];
    const float* bo   = (const float*)d_in[16];
    const float* W1   = (const float*)d_in[17];
    const float* b1   = (const float*)d_in[18];
    const float* W2   = (const float*)d_in[19];
    const float* b2   = (const float*)d_in[20];
    const float* g1   = (const float*)d_in[21];
    const float* be1  = (const float*)d_in[22];
    const float* g2   = (const float*)d_in[23];
    const float* be2  = (const float*)d_in[24];

    char* ws = (char*)d_ws;
    size_t off = 0;
    auto walloc = [&](size_t bytes) -> void* {
        void* p = ws + off;
        off += (bytes + 255) & ~(size_t)255;
        return p;
    };
    u16*   xb     = (u16*)  walloc((size_t)BS * 1024 * 2);
    u16*   Wt     = (u16*)  walloc((size_t)4096 * 1024 * 2);   // reused for every weight
    float* tmp    = (float*)walloc((size_t)BS * 1024 * 4);     // reused fp32 GEMM out
    u16*   qrb    = (u16*)  walloc((size_t)BS * 1024 * 2);
    u16*   qnb    = (u16*)  walloc((size_t)BS * 1024 * 2);     // holds -qi
    u16*   krb    = (u16*)  walloc((size_t)BS * 1024 * 2);
    u16*   kib    = (u16*)  walloc((size_t)BS * 1024 * 2);
    u16*   vTb    = (u16*)  walloc((size_t)BS * 1024 * 2);
    u16*   headsb = (u16*)  walloc((size_t)BS * 1024 * 2);
    float* h1     = (float*)walloc((size_t)BS * 1024 * 4);
    u16*   h1b    = (u16*)  walloc((size_t)BS * 1024 * 2);
    u16*   fb     = (u16*)  walloc((size_t)BS * FFD * 2);

    const float scale = 0.125f;  // 1/sqrt(64)
    dim3 gblk(32, 8);
    int n_tok = BS * 1024;
    int gEW = (n_tok + 255) / 256;
    int gHW = (1024 * HH + 255) / 256;

    k_f2bf<<<gEW, 256, 0, stream>>>(x, xb, n_tok);

    // --- QKV projections (weight buffer reused; stream order serializes) ---
    k_headw_t<<<gHW, 256, 0, stream>>>(Wq_r, Wt);
    k_gemm<<<dim3(16, 16), gblk, 0, stream>>>(xb, Wt, bq_r, tmp, BS, 1024, 1024, 0);
    k_epi_q<<<gEW, 256, 0, stream>>>(tmp, pqr, qrb, scale, 0);

    k_headw_t<<<gHW, 256, 0, stream>>>(Wq_i, Wt);
    k_gemm<<<dim3(16, 16), gblk, 0, stream>>>(xb, Wt, bq_i, tmp, BS, 1024, 1024, 0);
    k_epi_q<<<gEW, 256, 0, stream>>>(tmp, pqi, qnb, scale, 1);   // negated

    k_headw_t<<<gHW, 256, 0, stream>>>(Wk_r, Wt);
    k_gemm<<<dim3(16, 16), gblk, 0, stream>>>(xb, Wt, bk_r, tmp, BS, 1024, 1024, 0);
    k_epi_k<<<gEW, 256, 0, stream>>>(tmp, pkr, krb);

    k_headw_t<<<gHW, 256, 0, stream>>>(Wk_i, Wt);
    k_gemm<<<dim3(16, 16), gblk, 0, stream>>>(xb, Wt, bk_i, tmp, BS, 1024, 1024, 0);
    k_epi_k<<<gEW, 256, 0, stream>>>(tmp, pki, kib);

    k_headw_t<<<gHW, 256, 0, stream>>>(Wv, Wt);
    k_gemm<<<dim3(16, 16), gblk, 0, stream>>>(xb, Wt, bv, tmp, BS, 1024, 1024, 0);
    k_epi_v<<<gEW, 256, 0, stream>>>(tmp, vTb);

    // --- fused attention (4096 waves, 8 per block) ---
    k_flash<<<512, gblk, 0, stream>>>(qrb, qnb, krb, kib, vTb, headsb, scale);

    // --- output projection + LN1 ---
    k_transpose_bf16<<<gHW, 256, 0, stream>>>(Wo, Wt, 1024, 1024);
    k_gemm<<<dim3(16, 16), gblk, 0, stream>>>(headsb, Wt, bo, tmp, BS, 1024, 1024, 0);
    k_ln<<<BS, 256, 0, stream>>>(x, tmp, g1, be1, h1, h1b, 1024);

    // --- FFN ---
    k_transpose_bf16<<<(1024 * FFD + 255) / 256, 256, 0, stream>>>(W1, Wt, 1024, FFD);
    k_gemm<<<dim3(FFD / 64, 16), gblk, 0, stream>>>(h1b, Wt, b1, fb, BS, FFD, 1024, 1);  // bias+relu+bf16

    k_transpose_bf16<<<(FFD * 1024 + 255) / 256, 256, 0, stream>>>(W2, Wt, FFD, 1024);
    k_gemm<<<dim3(16, 16), gblk, 0, stream>>>(fb, Wt, b2, tmp, BS, 1024, FFD, 0);

    k_ln<<<BS, 256, 0, stream>>>(h1, tmp, g2, be2, (float*)d_out, (u16*)nullptr, 1024);
}
